// GQAAttention_16758962389714
// MI455X (gfx1250) — matrix operand order repaired and checked by reading, not compiled, not run on hardware
//
#include <hip/hip_runtime.h>
#include <hip/hip_bf16.h>
#include <stdint.h>

// Problem constants (reference: D=1024, 16 heads, 4 KV groups, hd=64, T=2048)
#define D_MODEL  1024
#define KV_DIM   256
#define HEAD_DIM 64
#define N_HEADS  16
#define HPG      4
#define T_SEQ    2048
#define QBLK     128   // q rows per workgroup (8 waves x 16)
#define KPAD     72    // padded LDS row stride (64 + 8) to spread banks

typedef __attribute__((ext_vector_type(16))) __bf16 bf16x16;
typedef __attribute__((ext_vector_type(8)))  __bf16 bf16x8;
typedef __attribute__((ext_vector_type(8)))  float  f32x8;

static __device__ __forceinline__ __bf16 f2bf(float f) {
  return (__bf16)f;  // hardware f32->bf16 cvt (RNE)
}
static __device__ __forceinline__ __bf16 bflo(unsigned w) {
  return __builtin_bit_cast(__bf16, (unsigned short)(w & 0xffffu));
}
static __device__ __forceinline__ __bf16 bfhi(unsigned w) {
  return __builtin_bit_cast(__bf16, (unsigned short)(w >> 16));
}

// Per-lane async copy of 16B from global to LDS (ASYNCcnt-tracked, no VGPR
// data staging). LDS offset = low 32 bits of generic pointer (ISA: LDS
// aperture truncates to addr[31:0]).
static __device__ __forceinline__ void async_copy16(const __bf16* gsrc, __bf16* ldst) {
  unsigned       l = (unsigned)(uintptr_t)ldst;
  unsigned long long g = (unsigned long long)(uintptr_t)gsrc;
  asm volatile("global_load_async_to_lds_b128 %0, %1, off" :: "v"(l), "v"(g) : "memory");
}
static __device__ __forceinline__ void wait_async0() {
  asm volatile("s_wait_asynccnt 0x0" ::: "memory");
}

__global__ void f32_to_bf16_kernel(const float* __restrict__ src,
                                   __bf16* __restrict__ dst, int n) {
  int i = blockIdx.x * blockDim.x + threadIdx.x;
  if (i < n) dst[i] = f2bf(src[i]);
}

// C[M,N] = (A[M,K](bf16) * B[K,N](bf16) + bias) * oscale via
// v_wmma_f32_16x16x32_bf16. Each wave: 16x64 strip, A-fragment reused over
// 4 N-tiles; A pairs fetched as packed dwords, B rows as 32B vectors.
template <bool F32OUT>
__global__ void gemm_bf16_wmma(const __bf16* __restrict__ A,
                               const __bf16* __restrict__ Bm,
                               const float* __restrict__ bias,
                               void* __restrict__ outp,
                               int M, int N, int K, float oscale) {
  const int lane  = threadIdx.x & 31;
  const int wave  = threadIdx.x >> 5;
  const int gw    = blockIdx.x * (blockDim.x >> 5) + wave;
  const int tn    = N >> 6;
  const int tasks = (M >> 4) * tn;
  if (gw >= tasks) return;
  const int mt    = gw / tn;
  const int nt    = gw % tn;
  const int khalf = lane >> 4;
  const int l15   = lane & 15;
  const int arow  = mt * 16 + l15;        // A fragment: lane = M row
  const int brow  = l15 + (khalf << 4);   // unused; B fragment: lane = N column

  const unsigned* arow32 = (const unsigned*)(A + (size_t)arow * K);

  f32x8 acc[4] = {f32x8{}, f32x8{}, f32x8{}, f32x8{}};

  for (int k0 = 0; k0 < K; k0 += 32) {
    bf16x16 af;
#pragma unroll
    for (int v = 0; v < 8; ++v) {
      const int kb = k0 + ((v >> 2) << 4) + (khalf << 3) + ((v & 3) << 1);
      const unsigned w = arow32[kb >> 1];
      af[2 * v]     = bflo(w);
      af[2 * v + 1] = bfhi(w);
    }
    const __bf16* bp = Bm + (size_t)k0 * N + nt * 64 + l15;
    bf16x16 b0, b1, b2, b3;
#pragma unroll
    for (int e = 0; e < 16; ++e) {
      const int kk = ((e >> 3) << 4) + (khalf << 3) + (e & 7);
      const __bf16* q = bp + (size_t)kk * N;
      b0[e] = q[0]; b1[e] = q[16]; b2[e] = q[32]; b3[e] = q[48];
    }
    acc[0] = __builtin_amdgcn_wmma_f32_16x16x32_bf16(false, af, false, b0, (short)0, acc[0], false, false);
    acc[1] = __builtin_amdgcn_wmma_f32_16x16x32_bf16(false, af, false, b1, (short)0, acc[1], false, false);
    acc[2] = __builtin_amdgcn_wmma_f32_16x16x32_bf16(false, af, false, b2, (short)0, acc[2], false, false);
    acc[3] = __builtin_amdgcn_wmma_f32_16x16x32_bf16(false, af, false, b3, (short)0, acc[3], false, false);
  }

#pragma unroll
  for (int j = 0; j < 4; ++j) {
#pragma unroll
    for (int r = 0; r < 8; ++r) {
      const int row = mt * 16 + r + (khalf << 3);
      const int col = nt * 64 + j * 16 + l15;
      const float v = (acc[j][r] + bias[col]) * oscale;
      if (F32OUT) ((float*)outp)[(size_t)row * N + col] = v;
      else        ((__bf16*)outp)[(size_t)row * N + col] = f2bf(v);
    }
  }
}

// GQA flash attention, transposed-score formulation, async double-buffered
// K/V staging. Grid: (T/QBLK, N_HEADS, B); 256 threads = 8 waves, wave w owns
// q rows [qb*QBLK + w*16, +16). S' = K*Q^T so the P-fragment for P*V is built
// entirely in registers from the S' accumulator. Softmax scale pre-folded
// into Q. Only diagonal chunks run the causal-mask code.
__global__ void __launch_bounds__(256)
gqa_flash_attn_kernel(const __bf16* __restrict__ Q,
                      const __bf16* __restrict__ Kx,
                      const __bf16* __restrict__ V,
                      __bf16* __restrict__ Aout) {
  const int lane  = threadIdx.x & 31;
  const int wave  = threadIdx.x >> 5;
  const int l15   = lane & 15;
  const int khalf = lane >> 4;

  const int qb   = blockIdx.x;
  const int head = blockIdx.y;
  const int bz   = blockIdx.z;
  const int g    = head / HPG;

  const int rowBase = qb * QBLK + wave * 16;
  const __bf16* Qb = Q  + ((size_t)bz * T_SEQ) * D_MODEL + head * HEAD_DIM;
  const __bf16* Kb = Kx + ((size_t)bz * T_SEQ) * KV_DIM  + g * HEAD_DIM;
  const __bf16* Vb = V  + ((size_t)bz * T_SEQ) * KV_DIM  + g * HEAD_DIM;

  __shared__ __bf16 KtB[2][32 * KPAD];
  __shared__ __bf16 VtB[2][32 * KPAD];

  const int dimLane = l15 + (khalf << 4);  // unused; B fragments: lane -> N column
  const int srow = threadIdx.x >> 3;       // staging: 256 threads = 32 rows x 8 segs
  const int sseg = threadIdx.x & 7;

  // Q^T B-fragments (one-time gather): Qt(kk,n) = Q[rowBase+n][c*32+kk]
  bf16x16 qbf[2];
#pragma unroll
  for (int c = 0; c < 2; ++c)
#pragma unroll
    for (int e = 0; e < 16; ++e)
      qbf[c][e] = Qb[(size_t)(rowBase + l15) * D_MODEL + c * 32 + ((e >> 3) << 4) + (khalf << 3) + (e & 7)];

  f32x8 o[4] = {f32x8{}, f32x8{}, f32x8{}, f32x8{}};
  float mi = -1e30f, li = 0.0f;

  const int qRow     = rowBase + l15;      // this lane's q row (S' N index)
  const int kcEndBlk = qb * QBLK + QBLK;   // block-level causal bound

  // prologue: async-stage chunk 0 into buffer 0
  async_copy16(Kb + (size_t)srow * KV_DIM + sseg * 8, &KtB[0][srow * KPAD + sseg * 8]);
  async_copy16(Vb + (size_t)srow * KV_DIM + sseg * 8, &VtB[0][srow * KPAD + sseg * 8]);
  wait_async0();
  __syncthreads();

  int cur = 0;
  for (int kc = 0; kc < kcEndBlk; kc += 32) {
    // ---- async-stage next chunk into the other buffer (overlaps WMMAs)
    if (kc + 32 < kcEndBlk) {
      const size_t goff = (size_t)(kc + 32 + srow) * KV_DIM + sseg * 8;
      async_copy16(Kb + goff, &KtB[cur ^ 1][srow * KPAD + sseg * 8]);
      async_copy16(Vb + goff, &VtB[cur ^ 1][srow * KPAD + sseg * 8]);
    }

    if (kc <= rowBase + 15) {  // wave-uniform: chunk intersects this wave's rows
      const __bf16* Ktc = KtB[cur];
      const __bf16* Vtc = VtB[cur];
      // ---- S' = K * Q^T : two 16-key tiles, head-dim in 2 chunks of 32
      f32x8 s[2] = {f32x8{}, f32x8{}};
#pragma unroll
      for (int t = 0; t < 2; ++t) {
#pragma unroll
        for (int c = 0; c < 2; ++c) {
          bf16x16 af;  // A(m,k) = K[kc + t*16 + m][c*32 + k], lane = key row
          const unsigned* kr = (const unsigned*)(Ktc + (t * 16 + l15) * KPAD + c * 32);
#pragma unroll
          for (int v = 0; v < 8; ++v) {
            const int kb = ((v >> 2) << 4) + (khalf << 3) + ((v & 3) << 1);
            const unsigned w = kr[kb >> 1];
            af[2 * v]     = bflo(w);
            af[2 * v + 1] = bfhi(w);
          }
          s[t] = __builtin_amdgcn_wmma_f32_16x16x32_bf16(false, af, false, qbf[c],
                                                         (short)0, s[t], false, false);
        }
      }
      // ---- causal mask only on diagonal chunks (scale pre-folded into Q)
      if (kc + 31 > rowBase) {
#pragma unroll
        for (int t = 0; t < 2; ++t)
#pragma unroll
          for (int r = 0; r < 8; ++r) {
            const int key = kc + t * 16 + r + (khalf << 3);
            if (key > qRow) s[t][r] = -1e30f;
          }
      }
      // ---- online softmax: 16 in-lane values + one cross-half shuffle
      float mx = s[0][0];
#pragma unroll
      for (int t = 0; t < 2; ++t)
#pragma unroll
        for (int r = 0; r < 8; ++r) mx = fmaxf(mx, s[t][r]);
      mx = fmaxf(mx, __shfl_xor(mx, 16, 32));
      const float mnew  = fmaxf(mi, mx);
      const float alpha = __expf(mi - mnew);
      float rs = 0.0f;
#pragma unroll
      for (int t = 0; t < 2; ++t)
#pragma unroll
        for (int r = 0; r < 8; ++r) {
          const float p = __expf(s[t][r] - mnew);
          s[t][r] = p;
          rs += p;
        }
      rs += __shfl_xor(rs, 16, 32);
      li = li * alpha + rs;
      mi = mnew;
      // ---- rescale O (O rows q' = r + 8*khalf need alpha from lane q')
      float ar[8];
#pragma unroll
      for (int r = 0; r < 8; ++r) ar[r] = __shfl(alpha, r + (khalf << 3), 32);
#pragma unroll
      for (int j = 0; j < 4; ++j)
#pragma unroll
        for (int r = 0; r < 8; ++r) o[j][r] *= ar[r];
      // ---- P fragment directly from S' registers (layouts coincide)
      bf16x16 pf;
#pragma unroll
      for (int v = 0; v < 8; ++v) {
        pf[2 * v]     = f2bf(s[v >> 2][2 * (v & 3)]);
        pf[2 * v + 1] = f2bf(s[v >> 2][2 * (v & 3) + 1]);
      }
      // ---- O += P * V (V fragment: lane = dim column, keys in table order from LDS)
      const __bf16* vr = Vtc + l15;
#pragma unroll
      for (int j = 0; j < 4; ++j) {
        bf16x16 vf;
#pragma unroll
        for (int e = 0; e < 16; ++e)
          vf[e] = vr[(((e >> 3) << 4) + (khalf << 3) + (e & 7)) * KPAD + j * 16];
        o[j] = __builtin_amdgcn_wmma_f32_16x16x32_bf16(false, pf, false, vf,
                                                       (short)0, o[j], false, false);
      }
    }

    wait_async0();     // own async loads into next buffer landed
    __syncthreads();   // everyone done reading cur & staging next
    cur ^= 1;
  }

  // ---- epilogue: O / l (broadcast l from lane q' = r + 8*khalf), store bf16
  float lr[8];
#pragma unroll
  for (int r = 0; r < 8; ++r) lr[r] = __shfl(li, r + (khalf << 3), 32);
#pragma unroll
  for (int j = 0; j < 4; ++j)
#pragma unroll
    for (int r = 0; r < 8; ++r) {
      const int row = rowBase + r + (khalf << 3);
      const int col = head * HEAD_DIM + j * 16 + l15;
      Aout[((size_t)bz * T_SEQ + row) * D_MODEL + col] = f2bf(o[j][r] / lr[r]);
    }
}

extern "C" void kernel_launch(void* const* d_in, const int* in_sizes, int n_in,
                              void* d_out, int out_size, void* d_ws, size_t ws_size,
                              hipStream_t stream) {
  const float* x  = (const float*)d_in[0];
  const float* Wq = (const float*)d_in[1];
  const float* bq = (const float*)d_in[2];
  const float* Wk = (const float*)d_in[3];
  const float* bk = (const float*)d_in[4];
  const float* Wv = (const float*)d_in[5];
  const float* bv = (const float*)d_in[6];
  const float* Wo = (const float*)d_in[7];
  const float* bo = (const float*)d_in[8];
  float* out = (float*)d_out;

  const int BT = in_sizes[0] / D_MODEL;  // B*T = 4096
  const int Bb = BT / T_SEQ;             // B = 2

  // Workspace layout (bf16 staging): ~35 MB total
  char* w = (char*)d_ws;
  __bf16* X16  = (__bf16*)w; w += (size_t)BT * D_MODEL * 2;
  __bf16* Wq16 = (__bf16*)w; w += (size_t)D_MODEL * D_MODEL * 2;
  __bf16* Wk16 = (__bf16*)w; w += (size_t)D_MODEL * KV_DIM * 2;
  __bf16* Wv16 = (__bf16*)w; w += (size_t)D_MODEL * KV_DIM * 2;
  __bf16* Wo16 = (__bf16*)w; w += (size_t)D_MODEL * D_MODEL * 2;
  __bf16* Q16  = (__bf16*)w; w += (size_t)BT * D_MODEL * 2;
  __bf16* K16  = (__bf16*)w; w += (size_t)BT * KV_DIM * 2;
  __bf16* V16  = (__bf16*)w; w += (size_t)BT * KV_DIM * 2;
  __bf16* A16  = (__bf16*)w; w += (size_t)BT * D_MODEL * 2;

  auto cvt = [&](const float* s, __bf16* d, int n) {
    f32_to_bf16_kernel<<<(n + 255) / 256, 256, 0, stream>>>(s, d, n);
  };
  cvt(x,  X16,  BT * D_MODEL);
  cvt(Wq, Wq16, D_MODEL * D_MODEL);
  cvt(Wk, Wk16, D_MODEL * KV_DIM);
  cvt(Wv, Wv16, D_MODEL * KV_DIM);
  cvt(Wo, Wo16, D_MODEL * D_MODEL);

  // Q/K/V projections; softmax scale 1/sqrt(64) folded into Q
  const int tasksQ  = (BT / 16) * (D_MODEL / 64);  // 4096 waves
  const int tasksKV = (BT / 16) * (KV_DIM / 64);   // 1024 waves
  gemm_bf16_wmma<false><<<tasksQ / 8, 256, 0, stream>>>(X16, Wq16, bq, Q16, BT, D_MODEL, D_MODEL, 0.125f);
  gemm_bf16_wmma<false><<<tasksKV / 8, 256, 0, stream>>>(X16, Wk16, bk, K16, BT, KV_DIM, D_MODEL, 1.0f);
  gemm_bf16_wmma<false><<<tasksKV / 8, 256, 0, stream>>>(X16, Wv16, bv, V16, BT, KV_DIM, D_MODEL, 1.0f);

  // Causal GQA flash attention (async double-buffered K/V, transposed scores)
  dim3 ag(T_SEQ / QBLK, N_HEADS, Bb);
  gqa_flash_attn_kernel<<<ag, 256, 0, stream>>>(Q16, K16, V16, A16);

  // Output projection -> f32 d_out
  gemm_bf16_wmma<true><<<tasksQ / 8, 256, 0, stream>>>(A16, Wo16, bo, out, BT, D_MODEL, D_MODEL, 1.0f);
}
